// SimpleReasoningLayer_5188320494017
// MI455X (gfx1250) — matrix operand order repaired and checked by reading, not compiled, not run on hardware
//
#include <hip/hip_runtime.h>
#include <cstdint>
#include <cstddef>

// ---------------------------------------------------------------------------
// CDNA5 / gfx1250 wave32 WMMA implementation of SimpleReasoningLayer.
// B=2, T=2048, C=1024, H=16, D=64, CH=512.  All heavy GEMMs + attention use
// v_wmma_f32_16x16x32_bf16 (f32 accumulate).  GEMM LDS staging is double-
// buffered with global_load_async_to_lds_b128 (ASYNCcnt) so the async DMA of
// tile t+1 overlaps the WMMAs of tile t.
// ---------------------------------------------------------------------------

typedef __bf16 v16bf __attribute__((ext_vector_type(16)));
typedef float  v8f   __attribute__((ext_vector_type(8)));

struct alignas(16) U128 { unsigned int w[4]; };
struct Frag32 { U128 a, b; };

static __device__ __forceinline__ unsigned short f2bf(float f) {
    unsigned int u = __float_as_uint(f);
    unsigned int r = u + 0x7FFFu + ((u >> 16) & 1u);
    return (unsigned short)(r >> 16);
}

// Assemble a v16bf A/B fragment from two 16-byte chunks (ISA 7.12.2 layouts).
static __device__ __forceinline__ v16bf ld_frag(const unsigned short* p0,
                                                const unsigned short* p1) {
    Frag32 f;
    f.a = *reinterpret_cast<const U128*>(p0);
    f.b = *reinterpret_cast<const U128*>(p1);
    return __builtin_bit_cast(v16bf, f);
}

// Async bulk copy: 16 bytes global -> LDS, tracked with ASYNCcnt.
static __device__ __forceinline__ void async_b128(const void* gptr,
                                                  void* lptr) {
    unsigned lds = (unsigned)(uintptr_t)lptr;           // LDS aperture: low 32b
    unsigned long long ga = (unsigned long long)(uintptr_t)gptr;
    asm volatile("global_load_async_to_lds_b128 %0, %1, off"
                 :
                 : "v"(lds), "v"(ga)
                 : "memory");
}
static __device__ __forceinline__ void async_wait0() {
    asm volatile("s_wait_asynccnt 0x0" ::: "memory");
}

// ---------------------------------------------------------------------------
// f32 -> bf16 conversion
// ---------------------------------------------------------------------------
__global__ void k_f32_to_bf16(const float* __restrict__ in,
                              unsigned short* __restrict__ out, int n) {
    int i = blockIdx.x * 256 + threadIdx.x;
    if (i < n) out[i] = f2bf(in[i]);
}

// ---------------------------------------------------------------------------
// Tiled bf16 WMMA GEMM:  out[M,Nc] = A[M,K](bf16) @ B[K,Nc](bf16) + bias
// Block tile 128x64, 8 waves, each wave 32x32 (2x2 WMMA accums), K-step 64
// (8 WMMAs per wave per stage).  Double-buffered LDS: tile t+1 is staged with
// async-to-LDS copies (A) and a register transpose (B) while tile t computes.
// ---------------------------------------------------------------------------
template <bool HASF, bool HASB>
__global__ __launch_bounds__(256) void k_gemm_bf16(
    const unsigned short* __restrict__ A, const unsigned short* __restrict__ Bm,
    const float* __restrict__ bias, float* __restrict__ outF,
    unsigned short* __restrict__ outB, int M, int Nc, int K) {
    __shared__ unsigned short sA[2][128 * 64];   // [buf][row][k]   32 KB
    __shared__ unsigned short sBt[2][64 * 64];   // [buf][n][k]     16 KB

    const int tid = threadIdx.x;
    const int lane = tid & 31, wave = tid >> 5;
    const int hl = lane >> 4, lcol = lane & 15;
    const int wm = wave & 3, wn = wave >> 2;
    const int m0 = blockIdx.x * 128, n0 = blockIdx.y * 64;

    // per-thread staging coordinates
    const int arow = tid >> 1, acb = (tid & 1) * 32;   // A: 128 rows x 64 k
    const int bkr = tid >> 2, bnb = (tid & 3) * 16;    // B: 64 k-rows x 64 n

    auto stage = [&](int k0, int buf) {
        // A tile (128x64) via async global->LDS b128 copies
        const unsigned short* g = A + (size_t)(m0 + arow) * K + k0 + acb;
        unsigned short* l = &sA[buf][arow * 64 + acb];
#pragma unroll
        for (int q = 0; q < 4; ++q) async_b128(g + q * 8, l + q * 8);
        // B tile (64x64) transposed into sBt[buf][n][k]
        const U128* gb =
            reinterpret_cast<const U128*>(Bm + (size_t)(k0 + bkr) * Nc + n0 + bnb);
        U128 b0 = gb[0], b1 = gb[1];
        const unsigned short* bs0 = reinterpret_cast<const unsigned short*>(&b0);
        const unsigned short* bs1 = reinterpret_cast<const unsigned short*>(&b1);
#pragma unroll
        for (int ii = 0; ii < 8; ++ii) {
            sBt[buf][(bnb + ii) * 64 + bkr] = bs0[ii];
            sBt[buf][(bnb + 8 + ii) * 64 + bkr] = bs1[ii];
        }
    };

    v8f acc[2][2] = {};

    const int nt = K / 64;
    stage(0, 0);
    for (int t = 0; t < nt; ++t) {
        async_wait0();        // copies for tile t complete (issued last iter)
        __syncthreads();      // visible to all waves
        if (t + 1 < nt) stage((t + 1) * 64, (t + 1) & 1);  // overlap with WMMA
        if (t + 2 < nt) {     // prefetch two tiles ahead into L2
            __builtin_prefetch(A + (size_t)(m0 + arow) * K + (t + 2) * 64 + acb, 0, 0);
            __builtin_prefetch(Bm + (size_t)((t + 2) * 64 + bkr) * Nc + n0 + bnb, 0, 0);
        }
        const unsigned short* cA = sA[t & 1];
        const unsigned short* cB = sBt[t & 1];

#pragma unroll
        for (int kc = 0; kc < 2; ++kc) {
            v16bf af[2], bfv[2];
#pragma unroll
            for (int i = 0; i < 2; ++i) {
                int rowL = wm * 32 + i * 16 + lcol;
                af[i] = ld_frag(&cA[rowL * 64 + kc * 32 + hl * 8],
                                &cA[rowL * 64 + kc * 32 + 16 + hl * 8]);
            }
#pragma unroll
            for (int j = 0; j < 2; ++j) {
                int colL = wn * 32 + j * 16 + lcol;
                bfv[j] = ld_frag(&cB[colL * 64 + kc * 32 + hl * 8],
                                 &cB[colL * 64 + kc * 32 + 16 + hl * 8]);
            }
#pragma unroll
            for (int i = 0; i < 2; ++i)
#pragma unroll
                for (int j = 0; j < 2; ++j)
                    acc[i][j] = __builtin_amdgcn_wmma_f32_16x16x32_bf16(
                        false, af[i], false, bfv[j], (short)0, acc[i][j], false,
                        false);
        }
        __syncthreads();      // all reads of buf[t&1] done before restaging
    }

#pragma unroll
    for (int i = 0; i < 2; ++i) {
#pragma unroll
        for (int j = 0; j < 2; ++j) {
            int col = n0 + wn * 32 + j * 16 + lcol;
            float bv = bias[col];
#pragma unroll
            for (int r = 0; r < 8; ++r) {
                int row = m0 + wm * 32 + i * 16 + r + hl * 8;
                float v = acc[i][j][r] + bv;
                size_t idx = (size_t)row * Nc + col;
                if constexpr (HASF) outF[idx] = v;
                if constexpr (HASB) outB[idx] = f2bf(v);
            }
        }
    }
}

// ---------------------------------------------------------------------------
// Per-row LayerNorm + exact GELU (in-place f32, optional bf16 copy)
// ---------------------------------------------------------------------------
template <bool HASB>
__global__ __launch_bounds__(256) void k_ln_gelu(float* __restrict__ io,
                                                 const float* __restrict__ g,
                                                 const float* __restrict__ beta,
                                                 unsigned short* __restrict__ outB,
                                                 int Cn) {
    __shared__ float red[256];
    int row = blockIdx.x, tid = threadIdx.x;
    float* p = io + (size_t)row * Cn;
    float s = 0.f;
    for (int c = tid; c < Cn; c += 256) s += p[c];
    red[tid] = s; __syncthreads();
    for (int o = 128; o > 0; o >>= 1) { if (tid < o) red[tid] += red[tid + o]; __syncthreads(); }
    float mean = red[0] / Cn; __syncthreads();
    float s2 = 0.f;
    for (int c = tid; c < Cn; c += 256) { float d = p[c] - mean; s2 += d * d; }
    red[tid] = s2; __syncthreads();
    for (int o = 128; o > 0; o >>= 1) { if (tid < o) red[tid] += red[tid + o]; __syncthreads(); }
    float rstd = rsqrtf(red[0] / Cn + 1e-5f);
    for (int c = tid; c < Cn; c += 256) {
        float y = (p[c] - mean) * rstd * g[c] + beta[c];
        y = 0.5f * y * (1.0f + erff(y * 0.70710678118f));
        p[c] = y;
        if constexpr (HASB) outB[(size_t)row * Cn + c] = f2bf(y);
    }
}

// ---------------------------------------------------------------------------
// importance = clamp(sigmoid(h @ w2 + b2), 1e-6);  xi_bf16 = x * importance
// ---------------------------------------------------------------------------
__global__ __launch_bounds__(256) void k_imp_gate(
    const float* __restrict__ h, const float* __restrict__ w2,
    const float* __restrict__ b2, const float* __restrict__ x,
    unsigned short* __restrict__ xib, int CHn, int Cn) {
    __shared__ float red[256];
    __shared__ float sg;
    int row = blockIdx.x, tid = threadIdx.x;
    float s = 0.f;
    for (int c = tid; c < CHn; c += 256) s += h[(size_t)row * CHn + c] * w2[c];
    red[tid] = s; __syncthreads();
    for (int o = 128; o > 0; o >>= 1) { if (tid < o) red[tid] += red[tid + o]; __syncthreads(); }
    if (tid == 0) {
        float z = red[0] + b2[0];
        float sig = 1.0f / (1.0f + __expf(-z));
        sg = fmaxf(sig, 1e-6f);
    }
    __syncthreads();
    float gv = sg;
    for (int c = tid; c < Cn; c += 256)
        xib[(size_t)row * Cn + c] = f2bf(x[(size_t)row * Cn + c] * gv);
}

// ---------------------------------------------------------------------------
// mean over T:  out[b,c] = mean_t reasoned[b,t,c]
// ---------------------------------------------------------------------------
__global__ void k_mean_T(const float* __restrict__ r, float* __restrict__ out,
                         int Bn, int Tn, int Cn) {
    int idx = blockIdx.x * 256 + threadIdx.x;
    if (idx >= Bn * Cn) return;
    int b = idx / Cn, c = idx % Cn;
    float s = 0.f;
    for (int t = 0; t < Tn; ++t) s += r[((size_t)b * Tn + t) * Cn + c];
    out[idx] = s / Tn;
}

// ---------------------------------------------------------------------------
// temp_net: Linear(C,CH) -> LN -> GELU -> Linear(CH,H) -> softplus + 0.5
// one block per batch (tiny: B=2)
// ---------------------------------------------------------------------------
__global__ __launch_bounds__(256) void k_temp_net(
    const float* __restrict__ mean, const float* __restrict__ w1,
    const float* __restrict__ b1, const float* __restrict__ g,
    const float* __restrict__ beta, const float* __restrict__ w2,
    const float* __restrict__ b2, float* __restrict__ temp, int Cn, int CHn,
    int Hn) {
    __shared__ float h[512];
    __shared__ float red[256];
    int b = blockIdx.x, tid = threadIdx.x;
    const float* m = mean + (size_t)b * Cn;
    for (int j = tid; j < CHn; j += 256) {
        float s = b1[j];
        for (int c = 0; c < Cn; ++c) s += m[c] * w1[(size_t)c * CHn + j];
        h[j] = s;
    }
    __syncthreads();
    float s = 0.f;
    for (int j = tid; j < CHn; j += 256) s += h[j];
    red[tid] = s; __syncthreads();
    for (int o = 128; o > 0; o >>= 1) { if (tid < o) red[tid] += red[tid + o]; __syncthreads(); }
    float mu = red[0] / CHn; __syncthreads();
    s = 0.f;
    for (int j = tid; j < CHn; j += 256) { float d = h[j] - mu; s += d * d; }
    red[tid] = s; __syncthreads();
    for (int o = 128; o > 0; o >>= 1) { if (tid < o) red[tid] += red[tid + o]; __syncthreads(); }
    float rstd = rsqrtf(red[0] / CHn + 1e-5f);
    __syncthreads();
    for (int j = tid; j < CHn; j += 256) {
        float y = (h[j] - mu) * rstd * g[j] + beta[j];
        h[j] = 0.5f * y * (1.0f + erff(y * 0.70710678118f));
    }
    __syncthreads();
    if (tid < Hn) {
        float s2 = b2[tid];
        for (int j = 0; j < CHn; ++j) s2 += h[j] * w2[(size_t)j * Hn + tid];
        float sp = (s2 > 20.f) ? s2 : log1pf(__expf(s2));
        temp[b * Hn + tid] = sp + 0.5f;
    }
}

// ---------------------------------------------------------------------------
// Flash-style attention, one block = (b, h, 128 q rows), 8 waves x 16 rows.
// Streams 64-key tiles: K staged via async-to-LDS, V staged transposed,
// S = Q K^T (WMMA), online softmax (lane shuffles), P via LDS round-trip,
// O += P V (WMMA).  logits = (S*scale + maskbias) * temp.
// ---------------------------------------------------------------------------
__global__ __launch_bounds__(256) void k_attn(
    const unsigned short* __restrict__ qp, const unsigned short* __restrict__ kp,
    const unsigned short* __restrict__ vp, const int* __restrict__ mask,
    const float* __restrict__ temp, unsigned short* __restrict__ ob, int Bn,
    int Hn, int Tn, int Cn) {
    __shared__ unsigned short Kt[64 * 64];       // [key][d]
    __shared__ unsigned short Vt[64 * 64];       // transposed: [d][key]
    __shared__ float mb[128 * 64];               // additive mask bias
    __shared__ unsigned short pbuf[8 * 16 * 64]; // per-wave P (bf16)

    const int qblk = blockIdx.x, h = blockIdx.y, b = blockIdx.z;
    const int qbase = qblk * 128;
    const int tid = threadIdx.x, lane = tid & 31, wave = tid >> 5;
    const int hl = lane >> 4, lcol = lane & 15;

    const float tv = temp[b * Hn + h];
    const float sct = tv * 0.125f;  // temp / sqrt(64)

    // Q fragments for this wave's 16 rows (K dim = D = 64 -> two k-steps)
    const int qtok = b * Tn + qbase + wave * 16 + lcol;
    const unsigned short* qrow = qp + (size_t)qtok * Cn + h * 64;
    v16bf aq[2];
#pragma unroll
    for (int ks = 0; ks < 2; ++ks)
        aq[ks] = ld_frag(qrow + ks * 32 + hl * 8, qrow + ks * 32 + 16 + hl * 8);

    float mrun[8], lrun[8];
    v8f O[4] = {};
#pragma unroll
    for (int r = 0; r < 8; ++r) { mrun[r] = -3.0e38f; lrun[r] = 0.f; }

    const int srow = tid >> 2, scb = (tid & 3) * 16;  // staging coords

    for (int kb0 = 0; kb0 < Tn; kb0 += 64) {
        // ---- stage K tile via async global->LDS copies ----
        {
            const unsigned short* g =
                kp + (size_t)(b * Tn + kb0 + srow) * Cn + h * 64 + scb;
            unsigned short* l = &Kt[srow * 64 + scb];
            async_b128(g, l);
            async_b128(g + 8, l + 8);
        }
        // ---- stage V tile transposed ----
        {
            const U128* gvp = reinterpret_cast<const U128*>(
                vp + (size_t)(b * Tn + kb0 + srow) * Cn + h * 64 + scb);
            U128 v0v = gvp[0], v1v = gvp[1];
            const unsigned short* vs0 = reinterpret_cast<const unsigned short*>(&v0v);
            const unsigned short* vs1 = reinterpret_cast<const unsigned short*>(&v1v);
#pragma unroll
            for (int ii = 0; ii < 8; ++ii) {
                Vt[(scb + ii) * 64 + srow] = vs0[ii];
                Vt[(scb + 8 + ii) * 64 + srow] = vs1[ii];
            }
        }
        // ---- stage mask bias (128 q rows x 64 keys) ----
        {
            int qr = tid >> 1, kc0 = (tid & 1) * 32;
            const int* mp = mask + ((size_t)(b * Tn + qbase + qr)) * Tn + kb0 + kc0;
#pragma unroll
            for (int ii = 0; ii < 32; ++ii)
                mb[qr * 64 + kc0 + ii] = mp[ii] ? 0.0f : -1.0e9f;
        }
        if (kb0 + 64 < Tn) {
            __builtin_prefetch(kp + (size_t)(b * Tn + kb0 + 64 + srow) * Cn + h * 64 + scb, 0, 0);
            __builtin_prefetch(vp + (size_t)(b * Tn + kb0 + 64 + srow) * Cn + h * 64 + scb, 0, 0);
        }
        async_wait0();
        __syncthreads();

        // ---- S = Q K^T  (4 key n-tiles x 2 k-steps) ----
        v8f S[4];
#pragma unroll
        for (int j = 0; j < 4; ++j) {
            v8f s = {};
            int key = j * 16 + lcol;
#pragma unroll
            for (int ks = 0; ks < 2; ++ks) {
                v16bf bfv = ld_frag(&Kt[key * 64 + ks * 32 + hl * 8],
                                    &Kt[key * 64 + ks * 32 + 16 + hl * 8]);
                s = __builtin_amdgcn_wmma_f32_16x16x32_bf16(false, aq[ks], false,
                                                            bfv, (short)0, s,
                                                            false, false);
            }
            S[j] = s;
        }

        // ---- masked, temperature-scaled logits + row max ----
        float rmax[8];
#pragma unroll
        for (int r = 0; r < 8; ++r) rmax[r] = -3.0e38f;
#pragma unroll
        for (int j = 0; j < 4; ++j) {
#pragma unroll
            for (int r = 0; r < 8; ++r) {
                int qloc = wave * 16 + r + hl * 8;
                float l = S[j][r] * sct + mb[qloc * 64 + j * 16 + lcol] * tv;
                S[j][r] = l;
                rmax[r] = fmaxf(rmax[r], l);
            }
        }
#pragma unroll
        for (int r = 0; r < 8; ++r) {
#pragma unroll
            for (int off = 8; off > 0; off >>= 1)
                rmax[r] = fmaxf(rmax[r], __shfl_xor(rmax[r], off));
        }

        // ---- online softmax update ----
        float rsum[8];
#pragma unroll
        for (int r = 0; r < 8; ++r) {
            float mnew = fmaxf(mrun[r], rmax[r]);
            float f = __expf(mrun[r] - mnew);
            lrun[r] *= f;
            mrun[r] = mnew;
#pragma unroll
            for (int j = 0; j < 4; ++j) O[j][r] *= f;
            rsum[r] = 0.f;
        }
#pragma unroll
        for (int j = 0; j < 4; ++j) {
#pragma unroll
            for (int r = 0; r < 8; ++r) {
                float p = __expf(S[j][r] - mrun[r]);
                rsum[r] += p;
                pbuf[wave * 1024 + (r + hl * 8) * 64 + j * 16 + lcol] = f2bf(p);
            }
        }
#pragma unroll
        for (int r = 0; r < 8; ++r) {
#pragma unroll
            for (int off = 8; off > 0; off >>= 1) rsum[r] += __shfl_xor(rsum[r], off);
            lrun[r] += rsum[r];
        }
        __syncthreads();

        // ---- O += P V ----
#pragma unroll
        for (int ks = 0; ks < 2; ++ks) {
            v16bf ap = ld_frag(&pbuf[wave * 1024 + lcol * 64 + ks * 32 + hl * 8],
                               &pbuf[wave * 1024 + lcol * 64 + ks * 32 + 16 + hl * 8]);
#pragma unroll
            for (int j = 0; j < 4; ++j) {
                v16bf bfv = ld_frag(&Vt[(j * 16 + lcol) * 64 + ks * 32 + hl * 8],
                                    &Vt[(j * 16 + lcol) * 64 + ks * 32 + 16 + hl * 8]);
                O[j] = __builtin_amdgcn_wmma_f32_16x16x32_bf16(false, ap, false, bfv,
                                                               (short)0, O[j], false,
                                                               false);
            }
        }
        __syncthreads();
    }

    // ---- normalize and store (bf16, [B*T, C] layout) ----
#pragma unroll
    for (int r = 0; r < 8; ++r) {
        float inv = (lrun[r] > 0.f) ? 1.0f / lrun[r] : 0.f;
        int tok = b * Tn + qbase + wave * 16 + r + hl * 8;
#pragma unroll
        for (int j = 0; j < 4; ++j)
            ob[(size_t)tok * Cn + h * 64 + j * 16 + lcol] = f2bf(O[j][r] * inv);
    }
}

// ---------------------------------------------------------------------------
// out = LN(x + y) * g + b
// ---------------------------------------------------------------------------
__global__ __launch_bounds__(256) void k_res_ln(const float* __restrict__ x,
                                                const float* __restrict__ y,
                                                const float* __restrict__ g,
                                                const float* __restrict__ b,
                                                float* __restrict__ out, int Cn) {
    __shared__ float red[256];
    int row = blockIdx.x, tid = threadIdx.x;
    const float* px = x + (size_t)row * Cn;
    const float* py = y + (size_t)row * Cn;
    float s = 0.f;
    for (int c = tid; c < Cn; c += 256) s += px[c] + py[c];
    red[tid] = s; __syncthreads();
    for (int o = 128; o > 0; o >>= 1) { if (tid < o) red[tid] += red[tid + o]; __syncthreads(); }
    float mean = red[0] / Cn; __syncthreads();
    float s2 = 0.f;
    for (int c = tid; c < Cn; c += 256) {
        float d = px[c] + py[c] - mean;
        s2 += d * d;
    }
    red[tid] = s2; __syncthreads();
    for (int o = 128; o > 0; o >>= 1) { if (tid < o) red[tid] += red[tid + o]; __syncthreads(); }
    float rstd = rsqrtf(red[0] / Cn + 1e-5f);
    for (int c = tid; c < Cn; c += 256) {
        float v = (px[c] + py[c] - mean) * rstd * g[c] + b[c];
        out[(size_t)row * Cn + c] = v;
    }
}

// ---------------------------------------------------------------------------
// Host orchestration
// ---------------------------------------------------------------------------
extern "C" void kernel_launch(void* const* d_in, const int* in_sizes, int n_in,
                              void* d_out, int out_size, void* d_ws,
                              size_t ws_size, hipStream_t stream) {
    (void)in_sizes; (void)n_in; (void)out_size; (void)ws_size;
    const int Bv = 2, Tv = 2048, Cv = 1024, Hv = 16, CHv = 512;
    const int Nv = Bv * Tv;

    const float* x        = (const float*)d_in[0];
    const int*   mask     = (const int*)d_in[1];
    const float* imp_w1   = (const float*)d_in[2];
    const float* imp_b1   = (const float*)d_in[3];
    const float* imp_g    = (const float*)d_in[4];
    const float* imp_beta = (const float*)d_in[5];
    const float* imp_w2   = (const float*)d_in[6];
    const float* imp_b2   = (const float*)d_in[7];
    const float* rsn_w1   = (const float*)d_in[8];
    const float* rsn_b1   = (const float*)d_in[9];
    const float* rsn_g    = (const float*)d_in[10];
    const float* rsn_beta = (const float*)d_in[11];
    const float* rsn_w2   = (const float*)d_in[12];
    const float* rsn_b2   = (const float*)d_in[13];
    const float* q_w      = (const float*)d_in[14];
    const float* q_b      = (const float*)d_in[15];
    const float* k_w      = (const float*)d_in[16];
    const float* k_b      = (const float*)d_in[17];
    const float* v_w      = (const float*)d_in[18];
    const float* v_b      = (const float*)d_in[19];
    const float* o_w      = (const float*)d_in[20];
    const float* o_b      = (const float*)d_in[21];
    const float* tmp_w1   = (const float*)d_in[22];
    const float* tmp_b1   = (const float*)d_in[23];
    const float* tmp_g    = (const float*)d_in[24];
    const float* tmp_beta = (const float*)d_in[25];
    const float* tmp_w2   = (const float*)d_in[26];
    const float* tmp_b2   = (const float*)d_in[27];
    const float* norm_g   = (const float*)d_in[28];
    const float* norm_b   = (const float*)d_in[29];
    float* out = (float*)d_out;

    char* ws = (char*)d_ws;
    size_t off = 0;
    auto alloc = [&](size_t bytes) -> char* {
        char* p = ws + off;
        off = (off + bytes + 255) & ~(size_t)255;
        return p;
    };

    unsigned short* xb      = (unsigned short*)alloc((size_t)Nv * Cv * 2);
    unsigned short* wb_imp1 = (unsigned short*)alloc((size_t)Cv * CHv * 2);
    unsigned short* wb_rsn1 = (unsigned short*)alloc((size_t)Cv * Cv * 2);
    unsigned short* wb_rsn2 = (unsigned short*)alloc((size_t)Cv * Cv * 2);
    unsigned short* wb_q    = (unsigned short*)alloc((size_t)Cv * Cv * 2);
    unsigned short* wb_k    = (unsigned short*)alloc((size_t)Cv * Cv * 2);
    unsigned short* wb_v    = (unsigned short*)alloc((size_t)Cv * Cv * 2);
    unsigned short* wb_o    = (unsigned short*)alloc((size_t)Cv * Cv * 2);
    float*          t0      = (float*)alloc((size_t)Nv * CHv * 4);
    unsigned short* xib     = (unsigned short*)alloc((size_t)Nv * Cv * 2);
    float*          t1      = (float*)alloc((size_t)Nv * Cv * 4);
    unsigned short* t1b     = (unsigned short*)alloc((size_t)Nv * Cv * 2);
    float*          rsn     = (float*)alloc((size_t)Nv * Cv * 4);
    unsigned short* rsnb    = (unsigned short*)alloc((size_t)Nv * Cv * 2);
    unsigned short* qbuf    = (unsigned short*)alloc((size_t)Nv * Cv * 2);
    unsigned short* kbuf    = (unsigned short*)alloc((size_t)Nv * Cv * 2);
    unsigned short* vbuf    = (unsigned short*)alloc((size_t)Nv * Cv * 2);
    unsigned short* attnb   = (unsigned short*)alloc((size_t)Nv * Cv * 2);
    float*          projo   = (float*)alloc((size_t)Nv * Cv * 4);
    float*          meanR   = (float*)alloc((size_t)Bv * Cv * 4);
    float*          tempv   = (float*)alloc((size_t)Bv * Hv * 4);

    auto conv = [&](const float* src, unsigned short* dst, int n) {
        k_f32_to_bf16<<<(n + 255) / 256, 256, 0, stream>>>(src, dst, n);
    };
    conv(x, xb, Nv * Cv);
    conv(imp_w1, wb_imp1, Cv * CHv);
    conv(rsn_w1, wb_rsn1, Cv * Cv);
    conv(rsn_w2, wb_rsn2, Cv * Cv);
    conv(q_w, wb_q, Cv * Cv);
    conv(k_w, wb_k, Cv * Cv);
    conv(v_w, wb_v, Cv * Cv);
    conv(o_w, wb_o, Cv * Cv);

    // importance net
    k_gemm_bf16<true, false><<<dim3(Nv / 128, CHv / 64), 256, 0, stream>>>(
        xb, wb_imp1, imp_b1, t0, nullptr, Nv, CHv, Cv);
    k_ln_gelu<false><<<Nv, 256, 0, stream>>>(t0, imp_g, imp_beta, nullptr, CHv);
    k_imp_gate<<<Nv, 256, 0, stream>>>(t0, imp_w2, imp_b2, x, xib, CHv, Cv);

    // reasoning net
    k_gemm_bf16<true, false><<<dim3(Nv / 128, Cv / 64), 256, 0, stream>>>(
        xib, wb_rsn1, rsn_b1, t1, nullptr, Nv, Cv, Cv);
    k_ln_gelu<true><<<Nv, 256, 0, stream>>>(t1, rsn_g, rsn_beta, t1b, Cv);
    k_gemm_bf16<true, true><<<dim3(Nv / 128, Cv / 64), 256, 0, stream>>>(
        t1b, wb_rsn2, rsn_b2, rsn, rsnb, Nv, Cv, Cv);

    // temperature net
    k_mean_T<<<(Bv * Cv + 255) / 256, 256, 0, stream>>>(rsn, meanR, Bv, Tv, Cv);
    k_temp_net<<<Bv, 256, 0, stream>>>(meanR, tmp_w1, tmp_b1, tmp_g, tmp_beta,
                                       tmp_w2, tmp_b2, tempv, Cv, CHv, Hv);

    // q/k/v projections (bf16 outputs only)
    k_gemm_bf16<false, true><<<dim3(Nv / 128, Cv / 64), 256, 0, stream>>>(
        rsnb, wb_q, q_b, nullptr, qbuf, Nv, Cv, Cv);
    k_gemm_bf16<false, true><<<dim3(Nv / 128, Cv / 64), 256, 0, stream>>>(
        xib, wb_k, k_b, nullptr, kbuf, Nv, Cv, Cv);
    k_gemm_bf16<false, true><<<dim3(Nv / 128, Cv / 64), 256, 0, stream>>>(
        xb, wb_v, v_b, nullptr, vbuf, Nv, Cv, Cv);

    // attention
    k_attn<<<dim3(Tv / 128, Hv, Bv), 256, 0, stream>>>(qbuf, kbuf, vbuf, mask,
                                                       tempv, attnb, Bv, Hv, Tv,
                                                       Cv);

    // output projection + residual LayerNorm
    k_gemm_bf16<true, false><<<dim3(Nv / 128, Cv / 64), 256, 0, stream>>>(
        attnb, wb_o, o_b, projo, nullptr, Nv, Cv, Cv);
    k_res_ln<<<Nv, 256, 0, stream>>>(x, projo, norm_g, norm_b, out, Cv);
}
